// DeepSeekMoE_41171556500147
// MI455X (gfx1250) — matrix operand order repaired and checked by reading, not compiled, not run on hardware
//
#include <hip/hip_runtime.h>
#include <stdint.h>

// ---------------- problem constants ----------------
#define H_DIM 2048
#define D_FF  8192
#define NE    8
#define NS    2
#define T_TOK 4096          // BATCH * SEQ

// ---------------- GEMM tiling ----------------
#define BM 128
#define BN 128
#define BK 64
#define APAD 72             // A LDS row pitch (elems): 144B (= TDM pad: 128B row + 16B pad)
#define BKPAD 72            // B LDS row stride (elems) + 8-elem skew per 16 rows
#define BS_ELEMS (BN * BKPAD + 64)

typedef __bf16 bf16_t;
typedef __attribute__((ext_vector_type(16))) __bf16 v16bf;
typedef __attribute__((ext_vector_type(8)))  float  v8f;
typedef __attribute__((ext_vector_type(4)))  unsigned int   v4u;
typedef __attribute__((ext_vector_type(16))) unsigned short v16u;
typedef __attribute__((ext_vector_type(4)))  unsigned int   u32x4;
typedef __attribute__((ext_vector_type(8)))  int            i32x8;
typedef __attribute__((ext_vector_type(4)))  int            i32x4;

#if defined(__has_builtin)
#  if __has_builtin(__builtin_amdgcn_tensor_load_to_lds) && __has_builtin(__builtin_amdgcn_s_wait_tensorcnt)
#    define HAS_TDM 1
#  endif
#endif
#ifndef HAS_TDM
#  define HAS_TDM 0
#endif

struct TokRec { int e0, e1, p0, p1; float g0, g1; };

__device__ __forceinline__ unsigned short f2bf(float f) {
  union { float f; uint32_t u; } v; v.f = f;
  uint32_t u = v.u;
  return (unsigned short)((u + 0x7FFFu + ((u >> 16) & 1u)) >> 16);   // RNE
}

// ---------------- f32 -> bf16 bulk convert ----------------
__global__ __launch_bounds__(256) void cvt4_kernel(const float4* __restrict__ s,
                                                   ushort4* __restrict__ d, long n4) {
  long i  = (long)blockIdx.x * blockDim.x + threadIdx.x;
  long st = (long)gridDim.x * blockDim.x;
  for (; i < n4; i += st) {
    float4 v = s[i];
    ushort4 o;
    o.x = f2bf(v.x); o.y = f2bf(v.y); o.z = f2bf(v.z); o.w = f2bf(v.w);
    d[i] = o;
  }
}

// ---------------- router ----------------
__global__ __launch_bounds__(256) void init_kernel(int* cnt) {
  if (threadIdx.x < NE && blockIdx.x == 0) cnt[threadIdx.x] = 0;
}

__global__ __launch_bounds__(256)
void router_kernel(const float* __restrict__ x, const float* __restrict__ wg,
                   float* __restrict__ logits_out, int* __restrict__ cnt,
                   int* __restrict__ idxList, TokRec* __restrict__ rec) {
  int tok  = (int)((blockIdx.x * blockDim.x + threadIdx.x) >> 5);   // one wave32 per token
  int lane = threadIdx.x & 31;
  if (tok >= T_TOK) return;
  const float* xr = x + (long)tok * H_DIM;
  float acc[NE];
#pragma unroll
  for (int e = 0; e < NE; ++e) acc[e] = 0.f;
  for (int d = lane; d < H_DIM; d += 32) {
    float xv = xr[d];
    const float* wr = wg + (long)d * NE;
#pragma unroll
    for (int e = 0; e < NE; ++e) acc[e] += xv * wr[e];
  }
#pragma unroll
  for (int e = 0; e < NE; ++e) {
#pragma unroll
    for (int off = 16; off > 0; off >>= 1) acc[e] += __shfl_xor(acc[e], off, 32);
  }
  if (lane == 0) {
    float mx = acc[0];
#pragma unroll
    for (int e = 1; e < NE; ++e) mx = fmaxf(mx, acc[e]);
    float p[NE]; float s = 0.f;
#pragma unroll
    for (int e = 0; e < NE; ++e) { p[e] = __expf(acc[e] - mx); s += p[e]; }
    float inv = 1.f / s;
    int e0 = 0;
#pragma unroll
    for (int e = 1; e < NE; ++e) if (p[e] > p[e0]) e0 = e;     // first index wins ties
    int e1 = (e0 == 0) ? 1 : 0;
#pragma unroll
    for (int e = 0; e < NE; ++e) { if (e == e0) continue; if (p[e] > p[e1]) e1 = e; }
#pragma unroll
    for (int e = 0; e < NE; ++e) logits_out[(long)tok * NE + e] = acc[e];
    int p0 = atomicAdd(&cnt[e0], 1);
    int p1 = atomicAdd(&cnt[e1], 1);
    idxList[e0 * T_TOK + p0] = tok;
    idxList[e1 * T_TOK + p1] = tok;
    TokRec r; r.e0 = e0; r.e1 = e1; r.p0 = p0; r.p1 = p1;
    r.g0 = p[e0] * inv; r.g1 = p[e1] * inv;
    rec[tok] = r;
  }
}

__global__ void scan_kernel(const int* __restrict__ cnt, int* __restrict__ off) {
  if (threadIdx.x == 0 && blockIdx.x == 0) {
    int a = 0;
    for (int e = 0; e < NE; ++e) { off[e] = a; a += cnt[e]; }
  }
}

// ---------------- TDM: async A-tile load (64 cols x 128 rows, 2B elems) ----------------
// D# per CDNA5 ISA §8.3/8.4.  pad_interval=4 (32 DWORDs = one 128B row),
// pad_amount=3 (4 DWORDs = 16B)  ->  LDS pitch 144B == APAD elems.
// tensor_dim1=nrows gives hardware zero-fill for OOB rows.
__device__ __forceinline__ void tdm_load_a(const bf16_t* gsrc, const bf16_t* lds_dst,
                                           unsigned td0, unsigned td1, unsigned long stride_e) {
#if HAS_TDM
  unsigned long ga = (unsigned long)gsrc;
  unsigned lds = (unsigned)(unsigned long)lds_dst;
  u32x4 g0;
  g0[0] = 1u;                                                   // count=1 user descriptor
  g0[1] = lds;                                                  // lds_addr
  g0[2] = (unsigned)(ga & 0xFFFFFFFFu);                         // global_addr[31:0]
  g0[3] = (unsigned)((ga >> 32) & 0x01FFFFFFu) | 0x80000000u;   // global_addr[56:32] | type=2
  i32x8 g1;
  g1[0] = (int)((1u << 16) | (1u << 20) | (4u << 22) | (3u << 25)); // 2B elems, pad enabled
  g1[1] = (int)((td0 & 0xFFFFu) << 16);                             // tensor_dim0 lo (abar=0)
  g1[2] = (int)((td0 >> 16) | ((td1 & 0xFFFFu) << 16));             // dim0 hi | dim1 lo
  g1[3] = (int)((td1 >> 16) | (64u << 16));                         // dim1 hi | tile_dim0=BK
  g1[4] = 128;                                                      // tile_dim1=BM rows
  g1[5] = (int)(unsigned)(stride_e & 0xFFFFFFFFu);                  // tensor_dim0_stride lo
  g1[6] = (int)(unsigned)((stride_e >> 32) & 0xFFFFu);              // stride hi | dim1_stride=0
  g1[7] = 0;
  i32x4 z4 = (i32x4)0;
  i32x8 z8 = (i32x8)0;
  __builtin_amdgcn_tensor_load_to_lds(g0, g1, z4, z4, z8, 0);
#else
  (void)gsrc; (void)lds_dst; (void)td0; (void)td1; (void)stride_e;
#endif
}
__device__ __forceinline__ void tdm_wait() {
#if HAS_TDM
  __builtin_amdgcn_s_wait_tensorcnt(0);
#endif
}

// ---------------- WMMA fragment helpers (CDNA5 layouts, wave32) ----------------
__device__ __forceinline__ v16bf load_a_frag(const bf16_t* As, int m, int ks, int h) {
  const bf16_t* p = As + m * APAD + ks + h * 8;
  v16bf a;
#pragma unroll
  for (int i = 0; i < 8; ++i) { a[i] = p[i]; a[8 + i] = p[16 + i]; }
  return a;
}
__device__ __forceinline__ v16bf load_b_frag(const bf16_t* Bs, int n, int ks, int h) {
  const bf16_t* p = Bs + (n * BKPAD + (n >> 4) * 8 + ks + h * 8);
  v16bf b;
#pragma unroll
  for (int i = 0; i < 8; ++i) { b[i] = p[i]; b[8 + i] = p[16 + i]; }
  return b;
}

__device__ __forceinline__ void wave_mma(const bf16_t* As, const bf16_t* Bs,
                                         v8f acc[4][2], int mWave, int nWave, int lane) {
  int ln = lane & 15, h = lane >> 4;
#pragma unroll
  for (int ks = 0; ks < BK; ks += 32) {
    v16bf b0 = load_b_frag(Bs, nWave + ln, ks, h);
    v16bf b1 = load_b_frag(Bs, nWave + 16 + ln, ks, h);
#pragma unroll
    for (int fm = 0; fm < 4; ++fm) {
      v16bf a = load_a_frag(As, mWave + fm * 16 + ln, ks, h);
      acc[fm][0] = __builtin_amdgcn_wmma_f32_16x16x32_bf16(false, a, false, b0, (short)0,
                                                           acc[fm][0], false, false);
      acc[fm][1] = __builtin_amdgcn_wmma_f32_16x16x32_bf16(false, a, false, b1, (short)0,
                                                           acc[fm][1], false, false);
    }
  }
}

// ---------------- staging helpers (256 threads/block, BK=64) ----------------
__device__ __forceinline__ void a_load(const bf16_t* gA, long lda, int tid, v4u r[4]) {
  int rr = tid >> 1, c0 = (tid & 1) * 32;
  const v4u* s = (const v4u*)(gA + (long)rr * lda + c0);
  r[0] = s[0]; r[1] = s[1]; r[2] = s[2]; r[3] = s[3];
}
__device__ __forceinline__ void a_load_gather(const bf16_t* xb, int gtok, int kt, int tid, v4u r[4]) {
  int c0 = (tid & 1) * 32;
  if (gtok >= 0) {
    const v4u* s = (const v4u*)(xb + (long)gtok * H_DIM + kt + c0);
    r[0] = s[0]; r[1] = s[1]; r[2] = s[2]; r[3] = s[3];
  } else {
    v4u z = (v4u)0; r[0] = z; r[1] = z; r[2] = z; r[3] = z;
  }
}
__device__ __forceinline__ void a_store(bf16_t* As, int tid, const v4u r[4]) {
  int rr = tid >> 1, c0 = (tid & 1) * 32;
  v4u* d = (v4u*)(As + rr * APAD + c0);
  d[0] = r[0]; d[1] = r[1]; d[2] = r[2]; d[3] = r[3];
}
__device__ __forceinline__ void b_load(const bf16_t* gB, long ldb, int tid, v16u& r0, v16u& r1) {
  int kk = (tid >> 3) * 2, c0 = (tid & 7) * 16;
  r0 = *(const v16u*)(gB + (long)kk * ldb + c0);
  r1 = *(const v16u*)(gB + (long)(kk + 1) * ldb + c0);
}
__device__ __forceinline__ void b_store(bf16_t* Bs, int tid, v16u r0, v16u r1) {
  int kk = (tid >> 3) * 2, c0 = (tid & 7) * 16;
#pragma unroll
  for (int i = 0; i < 16; ++i) {
    unsigned p = (unsigned)r0[i] | ((unsigned)r1[i] << 16);
    int n = c0 + i;
    *(unsigned*)(Bs + n * BKPAD + (n >> 4) * 8 + kk) = p;
  }
}

// ---------------- shared-expert GEMM1 ----------------
__global__ __launch_bounds__(256)
void gemm1_shared(const bf16_t* __restrict__ xb, const bf16_t* __restrict__ w1b,
                  const float* __restrict__ b1, unsigned short* __restrict__ hb) {
  int mB = blockIdx.x * BM, nB = blockIdx.y * BN, s = blockIdx.z;
  const bf16_t* gA = xb + (long)mB * H_DIM;
  const bf16_t* gB = w1b + (long)s * H_DIM * D_FF + nB;
  __shared__ bf16_t As[2][BM * APAD];
  __shared__ bf16_t Bs[2][BS_ELEMS];
  int tid = threadIdx.x, lane = tid & 31;
  int mWave = ((tid >> 5) >> 2) * 64, nWave = ((tid >> 5) & 3) * 32;
  v8f acc[4][2] = {};
  v16u rb0, rb1;
#if HAS_TDM
  b_load(gB, D_FF, tid, rb0, rb1);
  if (tid < 32) tdm_load_a(gA, &As[0][0], 0x40000000u, 0x40000000u, H_DIM);
  b_store(&Bs[0][0], tid, rb0, rb1);
  if (tid < 32) tdm_wait();
#else
  v4u ra[4];
  a_load(gA, H_DIM, tid, ra);
  b_load(gB, D_FF, tid, rb0, rb1);
  a_store(&As[0][0], tid, ra);
  b_store(&Bs[0][0], tid, rb0, rb1);
#endif
  __syncthreads();
  for (int kt = 0; kt < H_DIM; kt += BK) {
    int cur = (kt >> 6) & 1;
    bool more = (kt + BK) < H_DIM;
    if (more) {
      b_load(gB + (long)(kt + BK) * D_FF, D_FF, tid, rb0, rb1);
#if HAS_TDM
      if (tid < 32) tdm_load_a(gA + kt + BK, &As[cur ^ 1][0], 0x40000000u, 0x40000000u, H_DIM);
#else
      a_load(gA + kt + BK, H_DIM, tid, ra);
#endif
    }
    wave_mma(&As[cur][0], &Bs[cur][0], acc, mWave, nWave, lane);
    if (more) {
      b_store(&Bs[cur ^ 1][0], tid, rb0, rb1);
#if !HAS_TDM
      a_store(&As[cur ^ 1][0], tid, ra);
#endif
    }
#if HAS_TDM
    if (more && tid < 32) tdm_wait();
#endif
    __syncthreads();
  }
  const float* bsrc = b1 + (long)s * D_FF + nB;
  unsigned short* C = hb + (long)mB * (2L * D_FF) + (long)s * D_FF + nB;
  int ln = lane & 15, h = lane >> 4;
#pragma unroll
  for (int fm = 0; fm < 4; ++fm)
#pragma unroll
    for (int fn = 0; fn < 2; ++fn) {
      int col = nWave + fn * 16 + ln;
      float bv = bsrc[col];
#pragma unroll
      for (int r = 0; r < 8; ++r) {
        int row = mWave + fm * 16 + h * 8 + r;
        float v = acc[fm][fn][r] + bv;
        C[(long)row * (2L * D_FF) + col] = f2bf(fmaxf(v, 0.f));
      }
    }
}

// ---------------- shared-expert GEMM2 ----------------
__global__ __launch_bounds__(256)
void gemm2_shared(const bf16_t* __restrict__ hb, const bf16_t* __restrict__ w2b,
                  const float* __restrict__ b2, float* __restrict__ so) {
  int mB = blockIdx.x * BM, nB = blockIdx.y * BN;
  const long K = 2L * D_FF;
  const bf16_t* gA = hb + (long)mB * K;
  const bf16_t* gB = w2b + nB;
  __shared__ bf16_t As[2][BM * APAD];
  __shared__ bf16_t Bs[2][BS_ELEMS];
  int tid = threadIdx.x, lane = tid & 31;
  int mWave = ((tid >> 5) >> 2) * 64, nWave = ((tid >> 5) & 3) * 32;
  v8f acc[4][2] = {};
  v16u rb0, rb1;
#if HAS_TDM
  b_load(gB, H_DIM, tid, rb0, rb1);
  if (tid < 32) tdm_load_a(gA, &As[0][0], 0x40000000u, 0x40000000u, (unsigned long)K);
  b_store(&Bs[0][0], tid, rb0, rb1);
  if (tid < 32) tdm_wait();
#else
  v4u ra[4];
  a_load(gA, K, tid, ra);
  b_load(gB, H_DIM, tid, rb0, rb1);
  a_store(&As[0][0], tid, ra);
  b_store(&Bs[0][0], tid, rb0, rb1);
#endif
  __syncthreads();
  for (long kt = 0; kt < K; kt += BK) {
    int cur = (int)(kt >> 6) & 1;
    bool more = (kt + BK) < K;
    if (more) {
      b_load(gB + (kt + BK) * H_DIM, H_DIM, tid, rb0, rb1);
#if HAS_TDM
      if (tid < 32) tdm_load_a(gA + kt + BK, &As[cur ^ 1][0], 0x40000000u, 0x40000000u, (unsigned long)K);
#else
      a_load(gA + kt + BK, K, tid, ra);
#endif
    }
    wave_mma(&As[cur][0], &Bs[cur][0], acc, mWave, nWave, lane);
    if (more) {
      b_store(&Bs[cur ^ 1][0], tid, rb0, rb1);
#if !HAS_TDM
      a_store(&As[cur ^ 1][0], tid, ra);
#endif
    }
#if HAS_TDM
    if (more && tid < 32) tdm_wait();
#endif
    __syncthreads();
  }
  float* C = so + (long)mB * H_DIM + nB;
  int ln = lane & 15, h = lane >> 4;
#pragma unroll
  for (int fm = 0; fm < 4; ++fm)
#pragma unroll
    for (int fn = 0; fn < 2; ++fn) {
      int col = nWave + fn * 16 + ln;
      float bv = b2[nB + col] + b2[H_DIM + nB + col];
#pragma unroll
      for (int r = 0; r < 8; ++r) {
        int row = mWave + fm * 16 + h * 8 + r;
        C[(long)row * H_DIM + col] = acc[fm][fn][r] + bv;
      }
    }
}

// ---------------- routed GEMM1 (gathered rows; register gather path) ----------------
__global__ __launch_bounds__(256)
void gemm1_routed(const bf16_t* __restrict__ xb, const bf16_t* __restrict__ w1e,
                  const float* __restrict__ b1e, const int* __restrict__ cnt,
                  const int* __restrict__ off, const int* __restrict__ idxList,
                  unsigned short* __restrict__ hr) {
  int e = blockIdx.z;
  int c = cnt[e];
  int mB = blockIdx.x * BM;
  if (mB >= c) return;
  int nrows = c - mB; if (nrows > BM) nrows = BM;
  int nB = blockIdx.y * BN;
  const int* rows = idxList + e * T_TOK + mB;
  const bf16_t* gB = w1e + (long)e * H_DIM * D_FF + nB;
  __shared__ bf16_t As[2][BM * APAD];
  __shared__ bf16_t Bs[2][BS_ELEMS];
  int tid = threadIdx.x, lane = tid & 31;
  int mWave = ((tid >> 5) >> 2) * 64, nWave = ((tid >> 5) & 3) * 32;
  int rr = tid >> 1;
  int gtok = (rr < nrows) ? rows[rr] : -1;
  v8f acc[4][2] = {};
  v4u ra[4]; v16u rb0, rb1;
  a_load_gather(xb, gtok, 0, tid, ra);
  b_load(gB, D_FF, tid, rb0, rb1);
  a_store(&As[0][0], tid, ra);
  b_store(&Bs[0][0], tid, rb0, rb1);
  __syncthreads();
  for (int kt = 0; kt < H_DIM; kt += BK) {
    int cur = (kt >> 6) & 1;
    bool more = (kt + BK) < H_DIM;
    if (more) {
      a_load_gather(xb, gtok, kt + BK, tid, ra);
      b_load(gB + (long)(kt + BK) * D_FF, D_FF, tid, rb0, rb1);
    }
    wave_mma(&As[cur][0], &Bs[cur][0], acc, mWave, nWave, lane);
    if (more) {
      a_store(&As[cur ^ 1][0], tid, ra);
      b_store(&Bs[cur ^ 1][0], tid, rb0, rb1);
    }
    __syncthreads();
  }
  const float* bsrc = b1e + (long)e * D_FF + nB;
  unsigned short* C = hr + (long)(off[e] + mB) * D_FF + nB;
  int ln = lane & 15, h = lane >> 4;
#pragma unroll
  for (int fm = 0; fm < 4; ++fm)
#pragma unroll
    for (int fn = 0; fn < 2; ++fn) {
      int col = nWave + fn * 16 + ln;
      float bv = bsrc[col];
#pragma unroll
      for (int r = 0; r < 8; ++r) {
        int row = mWave + fm * 16 + h * 8 + r;
        if (row < nrows) {
          float v = acc[fm][fn][r] + bv;
          C[(long)row * D_FF + col] = f2bf(fmaxf(v, 0.f));
        }
      }
    }
}

// ---------------- routed GEMM2 (TDM bounded A: hardware OOB zero-fill) ----------------
__global__ __launch_bounds__(256)
void gemm2_routed(const bf16_t* __restrict__ hr, const bf16_t* __restrict__ w2e,
                  const float* __restrict__ b2e, const int* __restrict__ cnt,
                  const int* __restrict__ off, float* __restrict__ yr) {
  int e = blockIdx.z;
  int c = cnt[e];
  int mB = blockIdx.x * BM;
  if (mB >= c) return;
  int nrows = c - mB; if (nrows > BM) nrows = BM;
  int nB = blockIdx.y * BN;
  int base = off[e] + mB;
  const bf16_t* gA = hr + (long)base * D_FF;
  const bf16_t* gB = w2e + (long)e * D_FF * H_DIM + nB;
  __shared__ bf16_t As[2][BM * APAD];
  __shared__ bf16_t Bs[2][BS_ELEMS];
  int tid = threadIdx.x, lane = tid & 31;
  int mWave = ((tid >> 5) >> 2) * 64, nWave = ((tid >> 5) & 3) * 32;
  v8f acc[4][2] = {};
  v16u rb0, rb1;
#if HAS_TDM
  b_load(gB, H_DIM, tid, rb0, rb1);
  if (tid < 32) tdm_load_a(gA, &As[0][0], 0x40000000u, (unsigned)nrows, D_FF);
  b_store(&Bs[0][0], tid, rb0, rb1);
  if (tid < 32) tdm_wait();
#else
  v4u ra[4];
  int rr = tid >> 1;
  {
    if (rr < nrows) a_load(gA, D_FF, tid, ra);
    else { v4u z = (v4u)0; ra[0] = z; ra[1] = z; ra[2] = z; ra[3] = z; }
  }
  b_load(gB, H_DIM, tid, rb0, rb1);
  a_store(&As[0][0], tid, ra);
  b_store(&Bs[0][0], tid, rb0, rb1);
#endif
  __syncthreads();
  for (int kt = 0; kt < D_FF; kt += BK) {
    int cur = (kt >> 6) & 1;
    bool more = (kt + BK) < D_FF;
    if (more) {
      b_load(gB + (long)(kt + BK) * H_DIM, H_DIM, tid, rb0, rb1);
#if HAS_TDM
      if (tid < 32) tdm_load_a(gA + kt + BK, &As[cur ^ 1][0], 0x40000000u, (unsigned)nrows, D_FF);
#else
      if (rr < nrows) a_load(gA + kt + BK, D_FF, tid, ra);
#endif
    }
    wave_mma(&As[cur][0], &Bs[cur][0], acc, mWave, nWave, lane);
    if (more) {
      b_store(&Bs[cur ^ 1][0], tid, rb0, rb1);
#if !HAS_TDM
      a_store(&As[cur ^ 1][0], tid, ra);
#endif
    }
#if HAS_TDM
    if (more && tid < 32) tdm_wait();
#endif
    __syncthreads();
  }
  const float* bsrc = b2e + (long)e * H_DIM + nB;
  float* C = yr + (long)base * H_DIM + nB;
  int ln = lane & 15, h = lane >> 4;
#pragma unroll
  for (int fm = 0; fm < 4; ++fm)
#pragma unroll
    for (int fn = 0; fn < 2; ++fn) {
      int col = nWave + fn * 16 + ln;
      float bv = bsrc[col];
#pragma unroll
      for (int r = 0; r < 8; ++r) {
        int row = mWave + fm * 16 + h * 8 + r;
        if (row < nrows) C[(long)row * H_DIM + col] = acc[fm][fn][r] + bv;
      }
    }
}

// ---------------- combine ----------------
__global__ __launch_bounds__(256)
void combine_kernel(const float* __restrict__ so, const float* __restrict__ yr,
                    const TokRec* __restrict__ rec, const int* __restrict__ off,
                    float* __restrict__ out) {
  long i = (long)blockIdx.x * blockDim.x + threadIdx.x;
  if (i >= (long)T_TOK * H_DIM) return;
  int t = (int)(i >> 11);
  int d = (int)(i & (H_DIM - 1));
  TokRec r = rec[t];
  long s0 = (long)(off[r.e0] + r.p0) * H_DIM + d;
  long s1 = (long)(off[r.e1] + r.p1) * H_DIM + d;
  out[i] = so[i] + r.g0 * yr[s0] + r.g1 * yr[s1];
}

// ---------------- host-side orchestration ----------------
static inline size_t alignup(size_t x) { return (x + 255) & ~(size_t)255; }

extern "C" void kernel_launch(void* const* d_in, const int* in_sizes, int n_in,
                              void* d_out, int out_size, void* d_ws, size_t ws_size,
                              hipStream_t stream) {
  const float* x     = (const float*)d_in[0];
  const float* ws_w1 = (const float*)d_in[1];
  const float* ws_b1 = (const float*)d_in[2];
  const float* ws_w2 = (const float*)d_in[3];
  const float* ws_b2 = (const float*)d_in[4];
  const float* we_w1 = (const float*)d_in[5];
  const float* we_b1 = (const float*)d_in[6];
  const float* we_w2 = (const float*)d_in[7];
  const float* we_b2 = (const float*)d_in[8];
  const float* wg    = (const float*)d_in[9];

  float* out    = (float*)d_out;
  float* logits = out + (size_t)T_TOK * H_DIM;

  char* ws = (char*)d_ws;
  size_t o = 0;
  bf16_t* xb  = (bf16_t*)(ws + o); o += alignup((size_t)T_TOK * H_DIM * 2);
  bf16_t* w1s = (bf16_t*)(ws + o); o += alignup((size_t)NS * H_DIM * D_FF * 2);
  bf16_t* w2s = (bf16_t*)(ws + o); o += alignup((size_t)NS * D_FF * H_DIM * 2);
  bf16_t* w1e = (bf16_t*)(ws + o); o += alignup((size_t)NE * H_DIM * D_FF * 2);
  bf16_t* w2e = (bf16_t*)(ws + o); o += alignup((size_t)NE * D_FF * H_DIM * 2);
  unsigned short* hs = (unsigned short*)(ws + o); o += alignup((size_t)T_TOK * 2 * D_FF * 2);
  float* so = (float*)(ws + o);   o += alignup((size_t)T_TOK * H_DIM * 4);
  unsigned short* hr = (unsigned short*)(ws + o); o += alignup((size_t)2 * T_TOK * D_FF * 2);
  float* yr = (float*)(ws + o);   o += alignup((size_t)2 * T_TOK * H_DIM * 4);
  int* cnt = (int*)(ws + o);      o += alignup(NE * sizeof(int));
  int* off = (int*)(ws + o);      o += alignup(NE * sizeof(int));
  int* idxList = (int*)(ws + o);  o += alignup((size_t)NE * T_TOK * sizeof(int));
  TokRec* rec = (TokRec*)(ws + o); o += alignup((size_t)T_TOK * sizeof(TokRec));
  (void)ws_size; (void)in_sizes; (void)n_in; (void)out_size;

  auto cvt = [&](const float* src, void* dst, long nelem) {
    long n4 = nelem >> 2;
    int blocks = (int)((n4 + 255) / 256); if (blocks > 8192) blocks = 8192;
    cvt4_kernel<<<blocks, 256, 0, stream>>>((const float4*)src, (ushort4*)dst, n4);
  };
  cvt(x,     xb,  (long)T_TOK * H_DIM);
  cvt(ws_w1, w1s, (long)NS * H_DIM * D_FF);
  cvt(ws_w2, w2s, (long)NS * D_FF * H_DIM);
  cvt(we_w1, w1e, (long)NE * H_DIM * D_FF);
  cvt(we_w2, w2e, (long)NE * D_FF * H_DIM);

  init_kernel<<<1, 32, 0, stream>>>(cnt);
  router_kernel<<<T_TOK / 8, 256, 0, stream>>>(x, wg, logits, cnt, idxList, rec);
  scan_kernel<<<1, 1, 0, stream>>>(cnt, off);

  gemm1_shared<<<dim3(T_TOK / BM, D_FF / BN, NS), 256, 0, stream>>>(xb, w1s, ws_b1, hs);
  gemm2_shared<<<dim3(T_TOK / BM, H_DIM / BN, 1), 256, 0, stream>>>((const bf16_t*)hs, w2s, ws_b2, so);

  gemm1_routed<<<dim3(T_TOK / BM, D_FF / BN, NE), 256, 0, stream>>>(
      xb, w1e, we_b1, cnt, off, idxList, hr);
  gemm2_routed<<<dim3(T_TOK / BM, H_DIM / BN, NE), 256, 0, stream>>>(
      (const bf16_t*)hr, w2e, we_b2, cnt, off, yr);

  long total = (long)T_TOK * H_DIM;
  combine_kernel<<<(unsigned)((total + 255) / 256), 256, 0, stream>>>(so, yr, rec, off, out);
}
